// TransformerBPBookLayer_85109071937683
// MI455X (gfx1250) — compile-verified, matrix operand order checked by reading, not run on hardware
//
#include <hip/hip_runtime.h>
#include <hip/hip_bf16.h>
#include <math.h>
#include <stdint.h>

// ---------------- types ----------------
typedef __bf16 bf16_t;
typedef __attribute__((ext_vector_type(16))) __bf16 v16bf;
typedef __attribute__((ext_vector_type(8)))  __bf16 v8bf;
typedef __attribute__((ext_vector_type(8)))  float   v8f;
typedef __attribute__((ext_vector_type(4)))  unsigned int v4u;
typedef __attribute__((ext_vector_type(8)))  int v8i;
typedef __attribute__((ext_vector_type(4)))  int v4i;

#define D_MODEL 512
#define N_HEADS 8
#define DHEAD   64
#define DFF_    2048
#define SLOTS_  1024
#define TOPK_   5
#define BATCH   8
#define SEQ     1024
#define NTOK    (BATCH*SEQ)          // 8192 tokens
#define TD      (3*D_MODEL)          // 1536 qkv row

// ---------------- WMMA fragment helpers (wave32, 16x16x32 bf16) ----------------
// A-frag (16x32, row-major source) and B-frag (32x16, source = row-major [N,K])
// share the same per-lane load pattern: lane l -> row (l&15), K offsets
// {(l>>4)*8 + 0..7} and {16 + (l>>4)*8 + 0..7}.  Two 16-byte loads per lane.
__device__ __forceinline__ v16bf load_frag(const bf16_t* p, int ld) {
  const int lane = threadIdx.x & 31;
  const bf16_t* q = p + (size_t)(lane & 15) * ld + (lane >> 4) * 8;
  v8bf lo = *(const v8bf*)(q);
  v8bf hi = *(const v8bf*)(q + 16);
  v16bf f;
#pragma unroll
  for (int i = 0; i < 8; ++i) { f[i] = lo[i]; f[i + 8] = hi[i]; }
  return f;
}

__device__ __forceinline__ v8f wmma_bf16(v16bf a, v16bf b, v8f c) {
  return __builtin_amdgcn_wmma_f32_16x16x32_bf16(false, a, false, b,
                                                 (short)0, c, false, false);
}

__device__ __forceinline__ float gelu_exact(float x) {
  return 0.5f * x * (1.0f + erff(x * 0.70710678118654752f));
}

// ---------------- Tensor Data Mover: 2D bf16 tile -> LDS ----------------
// Moves a [rows x cols] bf16 tile (row stride = stride_elems) into LDS with a
// 16B pad inserted after every 128B row (so cols must be 64 -> LDS stride 72).
// D# group0: count=1 | lds_addr | global_addr | type=2
// D# group1: data_size=2B, pad_enable, pad_interval=32dw, pad_amount=4dw,
//            tensor_dim0=cols, tensor_dim1=rows, tile_dim0=cols, tile_dim1=rows,
//            tensor_dim0_stride=stride_elems
__device__ __forceinline__ unsigned lds_off32(const void* p) {
  return (unsigned)(unsigned long long)(uintptr_t)p;  // low 32 bits = LDS offset
}

__device__ __forceinline__ void tdm_load_tile_bf16(unsigned lds_addr, const void* gptr,
                                                   int rows, int cols, int stride_elems) {
  unsigned long long ga = (unsigned long long)(uintptr_t)gptr;
  v4u g0;
  g0[0] = 1u;                                                 // count=1 (valid user D#)
  g0[1] = lds_addr;                                           // LDS byte address
  g0[2] = (unsigned)(ga & 0xffffffffu);                       // global_addr[31:0]
  g0[3] = (unsigned)((ga >> 32) & 0x01ffffffu) | 0x80000000u; // global_addr[56:32] | type=2
  v8i g1;
  g1[0] = (1 << 16) | (1 << 20) | (4 << 22) | (3 << 25);      // 2B, pad 4dw per 32dw
  g1[1] = (cols & 0xffff) << 16;                              // tensor_dim0[15:0]
  g1[2] = (rows & 0xffff) << 16;                              // dim0[31:16]=0 | tensor_dim1[15:0]
  g1[3] = (cols & 0xffff) << 16;                              // dim1[31:16]=0 | tile_dim0
  g1[4] = rows & 0xffff;                                      // tile_dim1 | tile_dim2=0
  g1[5] = stride_elems;                                       // tensor_dim0_stride[31:0]
  g1[6] = 0;                                                  // stride0[47:32]=0 | stride1 lo
  g1[7] = 0;
  v4i z4 = {0, 0, 0, 0};
  v8i z8 = {0, 0, 0, 0, 0, 0, 0, 0};
  // amdgpu-toolchain (clang-23) 6-arg form: (g0, g1, g2, g3, g4, cpol)
  __builtin_amdgcn_tensor_load_to_lds(g0, g1, z4, z4, z8, 0);
}

// ---------------- elementwise cast f32 -> bf16 ----------------
__global__ void __launch_bounds__(256) cast_bf16_k(const float* __restrict__ x,
                                                   bf16_t* __restrict__ y, int n) {
  int i = blockIdx.x * 256 + threadIdx.x;
  if (i < n) y[i] = (bf16_t)x[i];
}

// ---------------- LayerNorm (row = 512) -> bf16 ----------------
__global__ void __launch_bounds__(256) ln_cast_k(const float* __restrict__ x,
                                                 const float* __restrict__ g,
                                                 const float* __restrict__ b,
                                                 bf16_t* __restrict__ y) {
  const int row = blockIdx.x, tid = threadIdx.x;
  const float* xr = x + (size_t)row * D_MODEL;
  float v0 = xr[tid], v1 = xr[tid + 256];
  __shared__ float s1[256], s2[256];
  s1[tid] = v0 + v1;
  s2[tid] = v0 * v0 + v1 * v1;
  __syncthreads();
  for (int s = 128; s > 0; s >>= 1) {
    if (tid < s) { s1[tid] += s1[tid + s]; s2[tid] += s2[tid + s]; }
    __syncthreads();
  }
  float mu  = s1[0] * (1.0f / D_MODEL);
  float var = s2[0] * (1.0f / D_MODEL) - mu * mu;
  float rs  = rsqrtf(var + 1e-5f);
  bf16_t* yr = y + (size_t)row * D_MODEL;
  yr[tid]       = (bf16_t)((v0 - mu) * rs * g[tid]       + b[tid]);
  yr[tid + 256] = (bf16_t)((v1 - mu) * rs * g[tid + 256] + b[tid + 256]);
}

// ---------------- L2 normalize (row = 512) -> bf16 ----------------
__global__ void __launch_bounds__(256) l2norm_cast_k(const float* __restrict__ x,
                                                     bf16_t* __restrict__ y) {
  const int row = blockIdx.x, tid = threadIdx.x;
  const float* xr = x + (size_t)row * D_MODEL;
  float v0 = xr[tid], v1 = xr[tid + 256];
  __shared__ float s1[256];
  s1[tid] = v0 * v0 + v1 * v1;
  __syncthreads();
  for (int s = 128; s > 0; s >>= 1) {
    if (tid < s) s1[tid] += s1[tid + s];
    __syncthreads();
  }
  float inv = 1.0f / fmaxf(sqrtf(s1[0]), 1e-12f);
  bf16_t* yr = y + (size_t)row * D_MODEL;
  yr[tid]       = (bf16_t)(v0 * inv);
  yr[tid + 256] = (bf16_t)(v1 * inv);
}

// ---------------- tiled WMMA GEMM:  C[M,N] = A[M,K] @ W[N,K]^T (+bias) ----------------
// TDM double-buffered staging: wave 0 issues next-stage tensor_load_to_lds while
// all waves compute the current stage.
// MODE 0: store bf16      MODE 1: gelu -> bf16
// MODE 2: +residual, store f32        MODE 3: store f32
#define BM 128
#define BN 128
#define BK 64
#define LDT 72   // padded LDS row stride (bf16): 144B = 9*16B, conflict-free
template <int MODE>
__global__ void __launch_bounds__(256) gemm_bf16_k(
    const bf16_t* __restrict__ A, const bf16_t* __restrict__ W,
    const float* __restrict__ bias, const float* __restrict__ resid,
    float* __restrict__ outF, bf16_t* __restrict__ outB,
    int M, int N, int K) {
  __shared__ alignas(16) bf16_t As[2][BM * LDT];
  __shared__ alignas(16) bf16_t Bs[2][BM * LDT];
  const int tid  = threadIdx.x;
  const int lane = tid & 31, wave = tid >> 5;
  const int l16 = lane & 15, half = lane >> 4;
  const int m0 = blockIdx.y * BM, n0 = blockIdx.x * BN;
  const int wm = (wave >> 1) * 32;  // 4 waves along M (32 rows each)
  const int wn = (wave & 1) * 64;   // 2 waves along N (64 cols each)

  v8f acc[2][4];
#pragma unroll
  for (int i = 0; i < 2; ++i)
#pragma unroll
    for (int j = 0; j < 4; ++j)
#pragma unroll
      for (int r = 0; r < 8; ++r) acc[i][j][r] = 0.0f;

  const int nst = K / BK;
  if (wave == 0) {  // wave-uniform branch: TDM issued once per block
    tdm_load_tile_bf16(lds_off32(&As[0][0]), &A[(size_t)m0 * K], BM, BK, K);
    tdm_load_tile_bf16(lds_off32(&Bs[0][0]), &W[(size_t)n0 * K], BN, BK, K);
  }
  for (int s = 0; s < nst; ++s) {
    const int p = s & 1;
    if (wave == 0) {
      if (s + 1 < nst) {
        tdm_load_tile_bf16(lds_off32(&As[p ^ 1][0]), &A[(size_t)m0 * K + (s + 1) * BK], BM, BK, K);
        tdm_load_tile_bf16(lds_off32(&Bs[p ^ 1][0]), &W[(size_t)n0 * K + (s + 1) * BK], BN, BK, K);
        __builtin_amdgcn_s_wait_tensorcnt(2);  // in-order: stage s done, next may fly
      } else {
        __builtin_amdgcn_s_wait_tensorcnt(0);
      }
    }
    __syncthreads();
#pragma unroll
    for (int ks = 0; ks < 2; ++ks) {
      v16bf af[2], bfr[4];
#pragma unroll
      for (int i = 0; i < 2; ++i) af[i]  = load_frag(&As[p][(wm + i * 16) * LDT + ks * 32], LDT);
#pragma unroll
      for (int j = 0; j < 4; ++j) bfr[j] = load_frag(&Bs[p][(wn + j * 16) * LDT + ks * 32], LDT);
#pragma unroll
      for (int i = 0; i < 2; ++i)
#pragma unroll
        for (int j = 0; j < 4; ++j)
          acc[i][j] = wmma_bf16(af[i], bfr[j], acc[i][j]);
    }
    __syncthreads();  // all done reading buf p before it is refilled next round
  }

  // epilogue: C layout = lane (l&15) is column, VGPR r is row (half*8+r)
#pragma unroll
  for (int i = 0; i < 2; ++i) {
#pragma unroll
    for (int j = 0; j < 4; ++j) {
      const int n = n0 + wn + j * 16 + l16;
      const float bv = bias ? bias[n] : 0.0f;
#pragma unroll
      for (int r = 0; r < 8; ++r) {
        const int m = m0 + wm + i * 16 + half * 8 + r;
        const size_t off = (size_t)m * N + n;
        float v = acc[i][j][r] + bv;
        if (MODE == 0)      outB[off] = (bf16_t)v;
        else if (MODE == 1) outB[off] = (bf16_t)gelu_exact(v);
        else if (MODE == 2) outF[off] = v + resid[off];
        else                outF[off] = v;
      }
    }
  }
}

// ---------------- flash attention: per (qblock=128, head, batch) ----------------
__global__ void __launch_bounds__(256) attn_k(const bf16_t* __restrict__ qkv,
                                              bf16_t* __restrict__ ctx) {
  const int qb = blockIdx.x, h = blockIdx.y, b = blockIdx.z;
  const int tid = threadIdx.x, lane = tid & 31, wave = tid >> 5;
  const int l16 = lane & 15, half = lane >> 4;
  __shared__ alignas(16) bf16_t Kt[128 * LDT];   // [key][d], stride 72 (TDM-padded)
  __shared__ alignas(16) bf16_t Vt[64 * 136];    // [d][key], stride 136
  __shared__ alignas(16) bf16_t Pt[128 * 136];   // [q][key], stride 136
  const float SC = 0.125f;                       // 1/sqrt(64)
  const int qrow0 = b * SEQ + qb * 128 + wave * 16;  // this wave's 16 query tokens

  v16bf qf[2];
  qf[0] = load_frag(qkv + (size_t)qrow0 * TD + h * DHEAD,      TD);
  qf[1] = load_frag(qkv + (size_t)qrow0 * TD + h * DHEAD + 32, TD);

  float rowmax[8], rowsum[8];
  v8f cacc[4];
#pragma unroll
  for (int r = 0; r < 8; ++r) { rowmax[r] = -1e30f; rowsum[r] = 0.0f; }
#pragma unroll
  for (int dt = 0; dt < 4; ++dt)
#pragma unroll
    for (int r = 0; r < 8; ++r) cacc[dt][r] = 0.0f;

  for (int kc = 0; kc < SEQ / 128; ++kc) {
    const size_t krow = (size_t)(b * SEQ + kc * 128);
    // stage K chunk [128 keys x 64 d] via TDM (wave 0), V transpose by all waves
    if (wave == 0)
      tdm_load_tile_bf16(lds_off32(&Kt[0]),
                         &qkv[krow * TD + D_MODEL + h * DHEAD], 128, 64, TD);
    for (int c = tid; c < 2048; c += 256) {
      int key = c >> 4, d0 = (c & 15) * 4;
      const bf16_t* s = &qkv[(krow + key) * TD + 2 * D_MODEL + h * DHEAD + d0];
      Vt[(d0 + 0) * 136 + key] = s[0];
      Vt[(d0 + 1) * 136 + key] = s[1];
      Vt[(d0 + 2) * 136 + key] = s[2];
      Vt[(d0 + 3) * 136 + key] = s[3];
    }
    if (wave == 0) __builtin_amdgcn_s_wait_tensorcnt(0);
    __syncthreads();

    // scores: 16 q x 128 k  (8 C-frags), contraction over d=64 (2 WMMA steps)
    v8f sf[8];
#pragma unroll
    for (int kt = 0; kt < 8; ++kt) {
      v8f s;
#pragma unroll
      for (int r = 0; r < 8; ++r) s[r] = 0.0f;
      s = wmma_bf16(qf[0], load_frag(&Kt[(kt * 16) * LDT + 0],  LDT), s);
      s = wmma_bf16(qf[1], load_frag(&Kt[(kt * 16) * LDT + 32], LDT), s);
#pragma unroll
      for (int r = 0; r < 8; ++r) s[r] *= SC;
      sf[kt] = s;
    }

    // online softmax update (row = half*8 + r, reduce across 16 lanes)
#pragma unroll
    for (int r = 0; r < 8; ++r) {
      float m = sf[0][r];
#pragma unroll
      for (int kt = 1; kt < 8; ++kt) m = fmaxf(m, sf[kt][r]);
      m = fmaxf(m, __shfl_xor(m, 1, 32));
      m = fmaxf(m, __shfl_xor(m, 2, 32));
      m = fmaxf(m, __shfl_xor(m, 4, 32));
      m = fmaxf(m, __shfl_xor(m, 8, 32));
      float nm = fmaxf(rowmax[r], m);
      float corr = __expf(rowmax[r] - nm);
      rowmax[r] = nm;
      rowsum[r] *= corr;
#pragma unroll
      for (int dt = 0; dt < 4; ++dt) cacc[dt][r] *= corr;
    }

    // P = exp(s - rowmax) -> bf16 into LDS (wave-private rows)
#pragma unroll
    for (int kt = 0; kt < 8; ++kt) {
#pragma unroll
      for (int r = 0; r < 8; ++r) {
        float p = __expf(sf[kt][r] - rowmax[r]);
        rowsum[r] += p;
        Pt[(wave * 16 + half * 8 + r) * 136 + kt * 16 + l16] = (bf16_t)p;
      }
    }

    // ctx += P @ V : contraction over 128 keys (4 WMMA k-steps), n = 64 d (4 tiles)
#pragma unroll
    for (int ks = 0; ks < 4; ++ks) {
      v16bf pf = load_frag(&Pt[(wave * 16) * 136 + ks * 32], 136);
#pragma unroll
      for (int dt = 0; dt < 4; ++dt) {
        v16bf vf = load_frag(&Vt[(dt * 16) * 136 + ks * 32], 136);
        cacc[dt] = wmma_bf16(pf, vf, cacc[dt]);
      }
    }
    __syncthreads();
  }

  // finalize: divide by full row sum, write ctx[token][h*64 + d] as bf16
#pragma unroll
  for (int r = 0; r < 8; ++r) {
    float s = rowsum[r];
    s += __shfl_xor(s, 1, 32);
    s += __shfl_xor(s, 2, 32);
    s += __shfl_xor(s, 4, 32);
    s += __shfl_xor(s, 8, 32);
    float inv = 1.0f / s;
    const int m = qrow0 + half * 8 + r;
#pragma unroll
    for (int dt = 0; dt < 4; ++dt)
      ctx[(size_t)m * D_MODEL + h * DHEAD + dt * 16 + l16] = (bf16_t)(cacc[dt][r] * inv);
  }
}

// ---------------- top-5 over 1024 slots + softmax + gather + residual ----------------
__global__ void __launch_bounds__(256) topk_retrieve_k(const float* __restrict__ sim,
                                                       const float* __restrict__ memory,
                                                       const float* __restrict__ scale,
                                                       float* __restrict__ src2) {
  const int tok = blockIdx.x, tid = threadIdx.x;
  __shared__ float row[SLOTS_];
  __shared__ float rv[256];
  __shared__ int   ri[256];
  __shared__ float topv[TOPK_];
  __shared__ int   topi[TOPK_];
  __shared__ float wts[TOPK_];
  const float* sr = sim + (size_t)tok * SLOTS_;
  for (int j = tid; j < SLOTS_; j += 256) row[j] = sr[j];
  __syncthreads();

  for (int t = 0; t < TOPK_; ++t) {
    float v = -1e30f; int i = SLOTS_;
    for (int j = tid; j < SLOTS_; j += 256) {
      float rj = row[j];
      if (rj > v || (rj == v && j < i)) { v = rj; i = j; }
    }
    rv[tid] = v; ri[tid] = i;
    __syncthreads();
    for (int s = 128; s > 0; s >>= 1) {
      if (tid < s) {
        float v2 = rv[tid + s]; int i2 = ri[tid + s];
        if (v2 > rv[tid] || (v2 == rv[tid] && i2 < ri[tid])) { rv[tid] = v2; ri[tid] = i2; }
      }
      __syncthreads();
    }
    if (tid == 0) { topv[t] = rv[0]; topi[t] = ri[0]; row[ri[0]] = -1e30f; }
    __syncthreads();
  }

  if (tid == 0) {
    float mx = topv[0], s = 0.0f;
    float e[TOPK_];
    for (int t = 0; t < TOPK_; ++t) { e[t] = __expf(topv[t] - mx); s += e[t]; }
    float inv = 1.0f / s;
    for (int t = 0; t < TOPK_; ++t) wts[t] = e[t] * inv;
  }
  __syncthreads();

  const float sc = scale[0];
  for (int d = tid; d < D_MODEL; d += 256) {
    float a = 0.0f;
#pragma unroll
    for (int t = 0; t < TOPK_; ++t) a += wts[t] * memory[(size_t)topi[t] * D_MODEL + d];
    src2[(size_t)tok * D_MODEL + d] += sc * a;
  }
}

// ---------------- host-side orchestration ----------------
extern "C" void kernel_launch(void* const* d_in, const int* in_sizes, int n_in,
                              void* d_out, int out_size, void* d_ws, size_t ws_size,
                              hipStream_t stream) {
  const float* src        = (const float*)d_in[0];
  const float* in_proj_w  = (const float*)d_in[1];
  const float* in_proj_b  = (const float*)d_in[2];
  const float* out_proj_w = (const float*)d_in[3];
  const float* out_proj_b = (const float*)d_in[4];
  const float* memory     = (const float*)d_in[5];
  const float* rscale     = (const float*)d_in[6];
  const float* ln1_g      = (const float*)d_in[7];
  const float* ln1_b      = (const float*)d_in[8];
  const float* ln2_g      = (const float*)d_in[9];
  const float* ln2_b      = (const float*)d_in[10];
  const float* w1         = (const float*)d_in[11];
  const float* b1         = (const float*)d_in[12];
  const float* w2         = (const float*)d_in[13];
  const float* b2         = (const float*)d_in[14];
  float* out = (float*)d_out;

  char* ws = (char*)d_ws;
  size_t o = 0;
  auto carve = [&](size_t bytes) -> char* {
    char* p = ws + o;
    o = (o + bytes + 255) & ~(size_t)255;
    return p;
  };
  bf16_t* wqB   = (bf16_t*)carve((size_t)TD * D_MODEL * 2);       // 1536x512
  bf16_t* woB   = (bf16_t*)carve((size_t)D_MODEL * D_MODEL * 2);  // 512x512
  bf16_t* w1B   = (bf16_t*)carve((size_t)DFF_ * D_MODEL * 2);     // 2048x512
  bf16_t* w2B   = (bf16_t*)carve((size_t)D_MODEL * DFF_ * 2);     // 512x2048
  bf16_t* h1    = (bf16_t*)carve((size_t)NTOK * D_MODEL * 2);
  bf16_t* qkvB  = (bf16_t*)carve((size_t)NTOK * TD * 2);
  bf16_t* ctxB  = (bf16_t*)carve((size_t)NTOK * D_MODEL * 2);
  float*  src2  = (float*) carve((size_t)NTOK * D_MODEL * 4);
  bf16_t* xnB   = (bf16_t*)carve((size_t)NTOK * D_MODEL * 2);
  bf16_t* mnB   = (bf16_t*)carve((size_t)SLOTS_ * D_MODEL * 2);
  float*  sim   = (float*) carve((size_t)NTOK * SLOTS_ * 4);
  bf16_t* h2    = (bf16_t*)carve((size_t)NTOK * D_MODEL * 2);
  bf16_t* ffn1B = (bf16_t*)carve((size_t)NTOK * DFF_ * 2);
  (void)ws_size; (void)n_in; (void)in_sizes; (void)out_size;

  // 1) weight casts
  cast_bf16_k<<<(TD * D_MODEL) / 256, 256, 0, stream>>>(in_proj_w, wqB, TD * D_MODEL);
  cast_bf16_k<<<(D_MODEL * D_MODEL) / 256, 256, 0, stream>>>(out_proj_w, woB, D_MODEL * D_MODEL);
  cast_bf16_k<<<(DFF_ * D_MODEL) / 256, 256, 0, stream>>>(w1, w1B, DFF_ * D_MODEL);
  cast_bf16_k<<<(D_MODEL * DFF_) / 256, 256, 0, stream>>>(w2, w2B, D_MODEL * DFF_);

  // 2) LN1 -> QKV
  ln_cast_k<<<NTOK, 256, 0, stream>>>(src, ln1_g, ln1_b, h1);
  gemm_bf16_k<0><<<dim3(TD / BN, NTOK / BM), 256, 0, stream>>>(
      h1, wqB, in_proj_b, nullptr, nullptr, qkvB, NTOK, TD, D_MODEL);

  // 3) attention
  attn_k<<<dim3(SEQ / 128, N_HEADS, BATCH), 256, 0, stream>>>(qkvB, ctxB);

  // 4) out-proj (+src residual) -> src2
  gemm_bf16_k<2><<<dim3(D_MODEL / BN, NTOK / BM), 256, 0, stream>>>(
      ctxB, woB, out_proj_b, src, src2, nullptr, NTOK, D_MODEL, D_MODEL);

  // 5) BP-book memory: normalize, sim, top-k retrieve (in-place on src2)
  l2norm_cast_k<<<NTOK, 256, 0, stream>>>(src2, xnB);
  l2norm_cast_k<<<SLOTS_, 256, 0, stream>>>(memory, mnB);
  gemm_bf16_k<3><<<dim3(SLOTS_ / BN, NTOK / BM), 256, 0, stream>>>(
      xnB, mnB, nullptr, nullptr, sim, nullptr, NTOK, SLOTS_, D_MODEL);
  topk_retrieve_k<<<NTOK, 256, 0, stream>>>(sim, memory, rscale, src2);

  // 6) FFN: LN2 -> gelu(h@w1^T+b1) -> @w2^T+b2 + src2 -> d_out
  ln_cast_k<<<NTOK, 256, 0, stream>>>(src2, ln2_g, ln2_b, h2);
  gemm_bf16_k<1><<<dim3(DFF_ / BN, NTOK / BM), 256, 0, stream>>>(
      h2, w1B, b1, nullptr, nullptr, ffn1B, NTOK, DFF_, D_MODEL);
  gemm_bf16_k<2><<<dim3(D_MODEL / BN, NTOK / BM), 256, 0, stream>>>(
      ffn1B, w2B, b2, src2, out, nullptr, NTOK, D_MODEL, DFF_);
}
